// SOM_72473278153190
// MI455X (gfx1250) — matrix operand order repaired and checked by reading, not compiled, not run on hardware
//
#include <hip/hip_runtime.h>
#include <math.h>

// ---------------------------------------------------------------------------
// SOM BMU search for MI455X (gfx1250, wave32, WMMA).
//
// dists[b,k] = sqrt(max(||x_b||^2 - 2 x_b.w_k + ||w_k||^2
//                       + 2*eps*(sum x_b - sum w_k) + D*eps^2, 0))
// GEMM (34.4 GFLOP) runs on v_wmma_f32_16x16x32_bf16; exact fp32 row/col
// correction terms restore precision of the norm terms. bf16 x + Wt = 16MB,
// fully L2-resident (192MB L2) => compute-bound on the WMMA pipe.
// ---------------------------------------------------------------------------

typedef __attribute__((ext_vector_type(16))) __bf16 v16bf;
typedef __attribute__((ext_vector_type(8)))  __bf16 v8bf;
typedef __attribute__((ext_vector_type(8)))  float  v8f;

static constexpr int   Bn   = 4096;   // batch rows
static constexpr int   Dn   = 1024;   // feature dim (GEMM K)
static constexpr int   Kn   = 4096;   // codebook entries (GEMM N)
static constexpr float EPSF = 1e-6f;

union FragBF { v16bf v; v8bf h[2]; };

// --------------------------- kernel 1: row stats ---------------------------
// rowterm[b] = ||x_b||^2 + 2*eps*sum(x_b) + D*eps^2 ; also x -> bf16.
__global__ void som_row_stats(const float* __restrict__ x,
                              __bf16* __restrict__ xb,
                              float* __restrict__ rowterm) {
    __shared__ float s1[256], s2[256];
    const int b = blockIdx.x;
    const int t = threadIdx.x;
    const float* xr  = x  + (size_t)b * Dn;
    __bf16*      xbr = xb + (size_t)b * Dn;
    float sx = 0.f, x2 = 0.f;
    for (int i = 0; i < Dn; i += 256) {
        float v = xr[i + t];
        xbr[i + t] = (__bf16)v;
        sx += v;
        x2 += v * v;
    }
    s1[t] = sx; s2[t] = x2;
    __syncthreads();
    for (int s = 128; s > 0; s >>= 1) {
        if (t < s) { s1[t] += s1[t + s]; s2[t] += s2[t + s]; }
        __syncthreads();
    }
    if (t == 0)
        rowterm[b] = s2[0] + 2.f * EPSF * s1[0] + (float)Dn * EPSF * EPSF;
}

// --------------------------- kernel 2: col stats ---------------------------
// colterm[k] = ||w_k||^2 - 2*eps*sum(w_k). weight is [D,K]; threads stride k
// so every d-iteration is a fully coalesced row read.
__global__ void som_col_stats(const float* __restrict__ w,
                              float* __restrict__ colterm) {
    const int k = blockIdx.x * blockDim.x + threadIdx.x;
    float sw = 0.f, w2 = 0.f;
    for (int d = 0; d < Dn; ++d) {
        float v = w[(size_t)d * Kn + k];
        sw += v;
        w2 += v * v;
    }
    colterm[k] = w2 - 2.f * EPSF * sw;
}

// ------------------- kernel 3: transpose + convert W -----------------------
// Wt[k][d] = bf16(weight[d][k]) via 32x32 LDS tile (coalesced both sides),
// so per-lane B fragments become two contiguous 16B runs along d.
__global__ void som_wt(const float* __restrict__ w, __bf16* __restrict__ Wt) {
    __shared__ float tile[32][33];
    const int k0 = blockIdx.x * 32, d0 = blockIdx.y * 32;
    const int tx = threadIdx.x, ty = threadIdx.y;      // blockDim = (32, 8)
    #pragma unroll
    for (int i = 0; i < 4; ++i)
        tile[ty + 8 * i][tx] = w[(size_t)(d0 + ty + 8 * i) * Kn + (k0 + tx)];
    __syncthreads();
    #pragma unroll
    for (int i = 0; i < 4; ++i)
        Wt[(size_t)(k0 + ty + 8 * i) * Dn + (d0 + tx)] =
            (__bf16)tile[tx][ty + 8 * i];
}

// ------------------- kernel 4: fused WMMA GEMM + argmin --------------------
// One block per 16 rows of x; 8 waves sweep all 4096 codebook columns.
// Fragment layouts (ISA 7.12.2, wave32, 16-bit):
//   A 16x32: lanes 0-15 row=lane, K = {0..7, 16..23}; lanes 16-31 row=lane-16,
//            K = {8..15, 24..31}  -> two 16B ds_load_b128 from the LDS x tile.
//   B 32x16: lanes 0-15 col=lane, K = {0..7, 16..23}; lanes 16-31 col=lane-16,
//            K = {8..15, 24..31}  -> two 16B global_load_b128 from Wt (L2).
//   C 16x16: lanes 0-15 n=lane; VGPR r -> m = r + (lane>=16 ? 8 : 0).
__global__ __launch_bounds__(256) void som_bmu_gemm(
    const __bf16* __restrict__ xb,
    const __bf16* __restrict__ Wt,
    const float* __restrict__ rowterm,
    const float* __restrict__ colterm,
    const float* __restrict__ locations,
    float* __restrict__ out,          // [0,8192): bmu_loc, [8192]: loss, then idx
    float* __restrict__ rowmin)
{
    __shared__ __bf16 xs[16][Dn];     // 32 KB x tile (bf16)
    __shared__ float  rtS[16];
    __shared__ float  redv[8][16];
    __shared__ int    redi[8][16];

    const int tid   = threadIdx.x;
    const int wave  = tid >> 5;
    const int lane  = tid & 31;
    const int mtile = blockIdx.x * 16;

    // Stage the 16xD x tile (contiguous rows) into LDS with b128 copies.
    {
        const v8bf* src = (const v8bf*)(xb + (size_t)mtile * Dn);
        v8bf*       dst = (v8bf*)&xs[0][0];
        for (int i = tid; i < 16 * Dn / 8; i += 256) dst[i] = src[i];
        if (tid < 16) rtS[tid] = rowterm[mtile + tid];
    }
    __syncthreads();

    const int nlo  = lane & 15;       // N (and A-row) within 16-wide tile
    const int hi   = lane >> 4;
    const int selA = hi * 8;          // A K-offset for this half-wave
    const int selB = hi * 8;          // B K-offset for this half-wave

    float minv[8];
    int   mini[8];
    #pragma unroll
    for (int r = 0; r < 8; ++r) { minv[r] = 3.4e38f; mini[r] = 0; }

    for (int n0 = wave * 64; n0 < Kn; n0 += 64 * 8) {
        v8f acc[4] = {};
        const __bf16* wp[4];
        #pragma unroll
        for (int j = 0; j < 4; ++j)
            wp[j] = Wt + (size_t)(n0 + 16 * j + nlo) * Dn + selB;

        for (int kk = 0; kk < Dn; kk += 32) {
            FragBF a;
            a.h[0] = *(const v8bf*)&xs[nlo][kk + selA];        // ds_load_b128
            a.h[1] = *(const v8bf*)&xs[nlo][kk + selA + 16];
            #pragma unroll
            for (int j = 0; j < 4; ++j) {
                FragBF bm;
                bm.h[0] = *(const v8bf*)(wp[j] + kk);          // global_load_b128
                bm.h[1] = *(const v8bf*)(wp[j] + kk + 16);
                acc[j] = __builtin_amdgcn_wmma_f32_16x16x32_bf16(
                    false, a.v, false, bm.v, (short)0, acc[j], false, false);
            }
        }

        // Fused epilogue: d = sqrt(max(rowterm + colterm - 2*xw, 0)), argmin.
        #pragma unroll
        for (int j = 0; j < 4; ++j) {
            const int   n  = n0 + 16 * j + nlo;
            const float ct = colterm[n];
            #pragma unroll
            for (int r = 0; r < 8; ++r) {
                float d2 = rtS[r + selA] + ct - 2.0f * acc[j][r];
                float dd = sqrtf(fmaxf(d2, 0.0f));
                if (dd < minv[r]) { minv[r] = dd; mini[r] = n; }
            }
        }
    }

    // Butterfly min-reduce across the 16 lanes holding the same rows
    // (masks 1..8 never cross the 16-lane halves in wave32).
    #pragma unroll
    for (int m = 1; m <= 8; m <<= 1) {
        #pragma unroll
        for (int r = 0; r < 8; ++r) {
            float ov = __shfl_xor(minv[r], m, 32);
            int   oi = __shfl_xor(mini[r], m, 32);
            if (ov < minv[r] || (ov == minv[r] && oi < mini[r])) {
                minv[r] = ov; mini[r] = oi;
            }
        }
    }
    if (nlo == 0) {                    // lanes 0 (rows 0-7) and 16 (rows 8-15)
        #pragma unroll
        for (int r = 0; r < 8; ++r) {
            redv[wave][r + selA] = minv[r];
            redi[wave][r + selA] = mini[r];
        }
    }
    __syncthreads();

    if (tid < 16) {                    // one thread per output row
        float bv = redv[0][tid];
        int   bi = redi[0][tid];
        for (int w = 1; w < 8; ++w) {
            float ov = redv[w][tid];
            int   oi = redi[w][tid];
            if (ov < bv || (ov == bv && oi < bi)) { bv = ov; bi = oi; }
        }
        const int m = mtile + tid;
        out[2 * m]     = locations[2 * bi];       // bmu_locations [B,1,2]
        out[2 * m + 1] = locations[2 * bi + 1];
        out[2 * Bn + 1 + m] = (float)bi;          // bmu_indexes
        rowmin[m] = bv;                           // per-row min dist
    }
}

// ------------------------- kernel 5: loss reduce ---------------------------
__global__ void som_loss(const float* __restrict__ rowmin,
                         float* __restrict__ out) {
    __shared__ float s[256];
    const int t = threadIdx.x;
    float acc = 0.f;
    for (int i = t; i < Bn; i += 256) acc += rowmin[i];
    s[t] = acc;
    __syncthreads();
    for (int st = 128; st > 0; st >>= 1) {
        if (t < st) s[t] += s[t + st];
        __syncthreads();
    }
    if (t == 0) out[2 * Bn] = s[0] / (float)Bn;
}

// ---------------------------------------------------------------------------
extern "C" void kernel_launch(void* const* d_in, const int* in_sizes, int n_in,
                              void* d_out, int out_size, void* d_ws, size_t ws_size,
                              hipStream_t stream) {
    const float* x   = (const float*)d_in[0];   // [4096,1024]
    const float* wgt = (const float*)d_in[1];   // [1024,4096]
    const float* loc = (const float*)d_in[2];   // [4096,2]
    float* out = (float*)d_out;                 // 8192 + 1 + 4096 floats

    // Workspace layout (~16.05 MB)
    char* ws = (char*)d_ws;
    __bf16* xb = (__bf16*)ws;                                   // 8 MB
    __bf16* Wt = (__bf16*)(ws + (size_t)Bn * Dn * 2);           // 8 MB
    float* rowterm = (float*)(ws + 2 * (size_t)Bn * Dn * 2);    // 16 KB
    float* colterm = rowterm + Bn;                              // 16 KB
    float* rowmin  = colterm + Kn;                              // 16 KB

    som_row_stats<<<Bn, 256, 0, stream>>>(x, xb, rowterm);
    som_col_stats<<<Kn / 256, 256, 0, stream>>>(wgt, colterm);
    som_wt<<<dim3(Kn / 32, Dn / 32), dim3(32, 8), 0, stream>>>(wgt, Wt);
    som_bmu_gemm<<<Bn / 16, 256, 0, stream>>>(xb, Wt, rowterm, colterm, loc,
                                              out, rowmin);
    som_loss<<<1, 256, 0, stream>>>(rowmin, out);
}
